// DynamicRoutedSPT_30837865185697
// MI455X (gfx1250) — matrix operand order repaired and checked by reading, not compiled, not run on hardware
//
#include <hip/hip_runtime.h>
#include <hip/hip_bf16.h>
#include <math.h>

#define B_   32
#define V_   5
#define N_   196
#define C_   768
#define H_   768
#define NT16 13   // ceil(196/16)
#define NG   12   // 768 / 64 output-channel groups per row-tile

typedef _Float16 v16h __attribute__((ext_vector_type(16)));
typedef _Float16 h8   __attribute__((ext_vector_type(8)));
typedef float    v8f  __attribute__((ext_vector_type(8)));

__device__ __forceinline__ v16h cat16(h8 lo, h8 hi) {
  return __builtin_shufflevector(lo, hi, 0,1,2,3,4,5,6,7,8,9,10,11,12,13,14,15);
}

// ---------------------------------------------------------------------------
// Kernel 1: convert x -> f16, compute initial c = mean over views -> f16
// ---------------------------------------------------------------------------
__global__ void prep_x(const float* __restrict__ x,
                       _Float16* __restrict__ xh,
                       _Float16* __restrict__ ch) {
  size_t idx = (size_t)blockIdx.x * blockDim.x + threadIdx.x;
  size_t total = (size_t)B_ * N_ * C_;
  if (idx >= total) return;
  int cc = (int)(idx % C_);
  int n  = (int)((idx / C_) % N_);
  int b  = (int)(idx / ((size_t)N_ * C_));
  size_t base = (((size_t)b * V_) * N_ + n) * C_ + cc;
  float s = 0.f;
#pragma unroll
  for (int v = 0; v < V_; ++v) {
    float xv = x[base + (size_t)v * N_ * C_];
    xh[base + (size_t)v * N_ * C_] = (_Float16)xv;
    s += xv;
  }
  ch[idx] = (_Float16)(s * (1.0f / V_));
}

// ---------------------------------------------------------------------------
// Kernel 2: transpose + convert weights into [n_out][k] f16 layout
//   wv_t[v][d][c]  = w_views[v][c][d]
//   w1a_t[h][c]    = w1[c][h]        (top half: multiplies x)
//   w1b_t[h][c]    = w1[C+c][h]      (bottom half: multiplies c)
// ---------------------------------------------------------------------------
__global__ void prep_w(const float* __restrict__ wv, const float* __restrict__ w1,
                       _Float16* __restrict__ wv_t,
                       _Float16* __restrict__ w1a_t,
                       _Float16* __restrict__ w1b_t) {
  size_t idx = (size_t)blockIdx.x * blockDim.x + threadIdx.x;
  const size_t per = (size_t)C_ * C_;
  if (idx < (size_t)V_ * per) {
    int v = (int)(idx / per);
    size_t r = idx % per;
    int d = (int)(r / C_), cin = (int)(r % C_);
    wv_t[idx] = (_Float16)wv[(size_t)v * per + (size_t)cin * C_ + d];
  } else if (idx < (size_t)(V_ + 1) * per) {
    size_t r = idx - (size_t)V_ * per;
    int h = (int)(r / C_), cin = (int)(r % C_);
    w1a_t[r] = (_Float16)w1[(size_t)cin * H_ + h];
  } else if (idx < (size_t)(V_ + 2) * per) {
    size_t r = idx - (size_t)(V_ + 1) * per;
    int h = (int)(r / C_), cin = (int)(r % C_);
    w1b_t[r] = (_Float16)w1[(size_t)(C_ + cin) * H_ + h];
  }
}

// ---------------------------------------------------------------------------
// Kernel 3: WMMA GEMM.  A: [slabs * 196, 768] f16 row-major (row clamp at 196).
// W: pre-transposed [n_out][768] f16 (optionally per-view: stride per slab%V).
// Each wave computes a 16(row) x 64(col) tile via 4 accumulators; K in 32-steps.
// out: f16 or f32 per flag.
// ---------------------------------------------------------------------------
__global__ void gemm16(const _Float16* __restrict__ A,
                       const _Float16* __restrict__ W,
                       long viewStride, int slabs,
                       void* __restrict__ out, int out_f16) {
  int lane = threadIdx.x & 31;
  int wave = threadIdx.x >> 5;
  int task = blockIdx.x * 8 + wave;
  int ntasks = slabs * NT16 * NG;
  if (task >= ntasks) return;

  int ng   = task % NG;
  int mt   = task / NG;
  int slab = mt / NT16;
  int t    = mt % NT16;

  const _Float16* wt = W + (viewStride ? (long)(slab % V_) * viewStride : 0);

  // A-fragment addressing (contiguous K order used here, not the ISA table order): lane -> row m = lane%16,
  // lanes 0-15 hold K=0..15, lanes 16-31 hold K=16..31 (matches B).
  int mrow  = lane & 15;
  int trow0 = t * 16 + mrow;
  int ldrow = trow0 < N_ ? trow0 : (N_ - 1);
  const _Float16* arow = A + ((size_t)slab * N_ + ldrow) * C_;
  int koffA = (lane & 16) ? 16 : 0;

  // B-fragment addressing: lane -> col n = lane%16; lanes 0-15 K=0..15,
  // lanes 16-31 K=16..31 (packed pairs along K) -> contiguous in W[n][k].
  int colb  = ng * 64 + (lane & 15);
  int khalf = (lane & 16) ? 16 : 0;
  const _Float16* bp0 = wt + (size_t)(colb +  0) * C_ + khalf;
  const _Float16* bp1 = wt + (size_t)(colb + 16) * C_ + khalf;
  const _Float16* bp2 = wt + (size_t)(colb + 32) * C_ + khalf;
  const _Float16* bp3 = wt + (size_t)(colb + 48) * C_ + khalf;

  v8f acc0 = {}, acc1 = {}, acc2 = {}, acc3 = {};

  for (int k0 = 0; k0 < C_; k0 += 32) {
    __builtin_prefetch(arow + k0 + 64, 0, 0);
    h8 alo = *(const h8*)(arow + k0 + koffA);
    h8 ahi = *(const h8*)(arow + k0 + 8 + koffA);
    v16h af = cat16(alo, ahi);

    v16h bf;
    bf = cat16(*(const h8*)(bp0 + k0), *(const h8*)(bp0 + k0 + 8));
    acc0 = __builtin_amdgcn_wmma_f32_16x16x32_f16(false, af, false, bf, (short)0, acc0, false, false);
    bf = cat16(*(const h8*)(bp1 + k0), *(const h8*)(bp1 + k0 + 8));
    acc1 = __builtin_amdgcn_wmma_f32_16x16x32_f16(false, af, false, bf, (short)0, acc1, false, false);
    bf = cat16(*(const h8*)(bp2 + k0), *(const h8*)(bp2 + k0 + 8));
    acc2 = __builtin_amdgcn_wmma_f32_16x16x32_f16(false, af, false, bf, (short)0, acc2, false, false);
    bf = cat16(*(const h8*)(bp3 + k0), *(const h8*)(bp3 + k0 + 8));
    acc3 = __builtin_amdgcn_wmma_f32_16x16x32_f16(false, af, false, bf, (short)0, acc3, false, false);
  }

  // D layout (f32 16x16): VGPR r holds M = r (lanes 0-15) / 8+r (lanes 16-31),
  // N = lane%16.
  int rowhalf = (lane & 16) ? 8 : 0;
  int colbase = ng * 64 + (lane & 15);
  size_t orow_base = (size_t)slab * N_;
  if (out_f16) {
    _Float16* O = (_Float16*)out;
#pragma unroll
    for (int r = 0; r < 8; ++r) {
      int trow = t * 16 + rowhalf + r;
      if (trow < N_) {
        size_t off = (orow_base + trow) * C_ + colbase;
        O[off +  0] = (_Float16)acc0[r];
        O[off + 16] = (_Float16)acc1[r];
        O[off + 32] = (_Float16)acc2[r];
        O[off + 48] = (_Float16)acc3[r];
      }
    }
  } else {
    float* O = (float*)out;
#pragma unroll
    for (int r = 0; r < 8; ++r) {
      int trow = t * 16 + rowhalf + r;
      if (trow < N_) {
        size_t off = (orow_base + trow) * C_ + colbase;
        O[off +  0] = acc0[r];
        O[off + 16] = acc1[r];
        O[off + 32] = acc2[r];
        O[off + 48] = acc3[r];
      }
    }
  }
}

// ---------------------------------------------------------------------------
// Kernel 4: per (b,n): scores s_v = w2 . tanh(hx[b,v,n,:] + hc[b,n,:] + b1) + b2,
// softmax over V, c_new = sum_v a_v * proj[b,v,n,:]; last iter also writes
// fp32 c and r = 1 - H(a)/log V to d_out.
// ---------------------------------------------------------------------------
__global__ void iter_kernel(const _Float16* __restrict__ hx,
                            const float* __restrict__ hc,
                            const _Float16* __restrict__ proj,
                            const float* __restrict__ b1v,
                            const float* __restrict__ w2v,
                            const float* __restrict__ b2v,
                            _Float16* __restrict__ chbuf,
                            float* __restrict__ outc,
                            float* __restrict__ outr,
                            int last) {
  __shared__ float tsh[H_];
  __shared__ float sred[256];
  __shared__ float a_sh[8];

  int bn = blockIdx.x;
  int b = bn / N_;
  int n = bn % N_;
  int tid = threadIdx.x;

  const float* hcrow = hc + (size_t)bn * H_;
  for (int h = tid; h < H_; h += 256) tsh[h] = hcrow[h] + b1v[h];
  __syncthreads();

  for (int v = 0; v < V_; ++v) {
    const _Float16* hxrow = hx + (((size_t)b * V_ + v) * N_ + n) * (size_t)H_;
    float p = 0.f;
    for (int h = tid; h < H_; h += 256)
      p += tanhf((float)hxrow[h] + tsh[h]) * w2v[h];
    sred[tid] = p;
    __syncthreads();
    for (int off = 128; off > 0; off >>= 1) {
      if (tid < off) sred[tid] += sred[tid + off];
      __syncthreads();
    }
    if (tid == 0) a_sh[v] = sred[0] + b2v[0];
    __syncthreads();
  }

  if (tid == 0) {
    float m = a_sh[0];
    for (int v = 1; v < V_; ++v) m = fmaxf(m, a_sh[v]);
    float e[V_], sum = 0.f;
    for (int v = 0; v < V_; ++v) { e[v] = expf(a_sh[v] - m); sum += e[v]; }
    float inv = 1.f / sum, ent = 0.f;
    for (int v = 0; v < V_; ++v) {
      float av = e[v] * inv;
      a_sh[v] = av;
      ent -= av * logf(av + 1e-8f);
    }
    a_sh[5] = 1.f - ent / logf((float)V_);
  }
  __syncthreads();

  float av[V_];
#pragma unroll
  for (int v = 0; v < V_; ++v) av[v] = a_sh[v];

  for (int ch = tid; ch < C_; ch += 256) {
    float acc = 0.f;
#pragma unroll
    for (int v = 0; v < V_; ++v)
      acc += av[v] * (float)proj[(((size_t)b * V_ + v) * N_ + n) * (size_t)C_ + ch];
    chbuf[(size_t)bn * C_ + ch] = (_Float16)acc;
    if (last) outc[(size_t)bn * C_ + ch] = acc;
  }
  if (last && tid == 0) outr[bn] = a_sh[5];
}

// ---------------------------------------------------------------------------
extern "C" void kernel_launch(void* const* d_in, const int* in_sizes, int n_in,
                              void* d_out, int out_size, void* d_ws, size_t ws_size,
                              hipStream_t stream) {
  (void)in_sizes; (void)n_in; (void)out_size; (void)ws_size;
  const float* x  = (const float*)d_in[0];
  const float* wv = (const float*)d_in[1];
  const float* w1 = (const float*)d_in[2];
  const float* b1 = (const float*)d_in[3];
  const float* w2 = (const float*)d_in[4];
  const float* b2 = (const float*)d_in[5];

  char* ws = (char*)d_ws;
  size_t off = 0;
  auto alloc = [&](size_t bytes) -> void* {
    void* p = ws + off;
    off += (bytes + 255) & ~(size_t)255;
    return p;
  };

  const size_t XEL = (size_t)B_ * V_ * N_ * C_;        // 31360 * 768
  _Float16* xh    = (_Float16*)alloc(XEL * 2);
  _Float16* projh = (_Float16*)alloc(XEL * 2);
  _Float16* hxh   = (_Float16*)alloc(XEL * 2);
  _Float16* wv_t  = (_Float16*)alloc((size_t)V_ * C_ * C_ * 2);
  _Float16* w1a_t = (_Float16*)alloc((size_t)C_ * C_ * 2);
  _Float16* w1b_t = (_Float16*)alloc((size_t)C_ * C_ * 2);
  _Float16* chbuf = (_Float16*)alloc((size_t)B_ * N_ * C_ * 2);
  float*    hc    = (float*)alloc((size_t)B_ * N_ * H_ * 4);

  float* outc = (float*)d_out;
  float* outr = outc + (size_t)B_ * N_ * C_;

  {
    size_t tot = (size_t)B_ * N_ * C_;
    prep_x<<<(int)((tot + 255) / 256), 256, 0, stream>>>(x, xh, chbuf);
  }
  {
    size_t tot = (size_t)(V_ + 2) * C_ * C_;
    prep_w<<<(int)((tot + 255) / 256), 256, 0, stream>>>(wv, w1, wv_t, w1a_t, w1b_t);
  }

  const int blocksBV = (B_ * V_ * NT16 * NG + 7) / 8;  // 3120
  const int blocksB  = (B_ * NT16 * NG + 7) / 8;       // 624

  // proj = x @ w_views[v]   (per-view weight)
  gemm16<<<blocksBV, 256, 0, stream>>>(xh, wv_t, (long)C_ * C_, B_ * V_, (void*)projh, 1);
  // hx = x @ w1_top         (iteration-invariant half of the score MLP)
  gemm16<<<blocksBV, 256, 0, stream>>>(xh, w1a_t, 0L, B_ * V_, (void*)hxh, 1);

  for (int it = 0; it < 3; ++it) {
    // hc = c @ w1_bot
    gemm16<<<blocksB, 256, 0, stream>>>(chbuf, w1b_t, 0L, B_, (void*)hc, 0);
    // scores + softmax + weighted view-sum -> new c (and outputs on last iter)
    iter_kernel<<<B_ * N_, 256, 0, stream>>>(hxh, hc, projh, b1, w2, b2,
                                             chbuf, outc, outr, it == 2 ? 1 : 0);
  }
}
